// GRU_SiLU_60756607369561
// MI455X (gfx1250) — compile-verified, matrix operand order checked by reading, not run on hardware
//
#include <hip/hip_runtime.h>
#include <hip/hip_bf16.h>

// GRU + SiLU gate for MI455X (gfx1250, wave32, WMMA).
// T=2048, B=32, I=512, H=512.
//
// Pipeline:
//   k_init       : zero grid-barrier counter, hb[0] = bf16(h0)
//   k_transpose  : weights fp32 (K,N) -> bf16 (N,K)   (x4: W, Wg_x, R, Wg_h)
//   k_cvt_x      : x fp32 -> bf16
//   k_inproj     : Wx = x@W + bx (fp32 scratch), Gx = x@Wg_x  (WMMA bf16->f32)
//   k_recurrent  : 32 persistent WGs, 1 grid barrier per timestep, WMMA for
//                  h@R and h_new@Wg_h. Weight fragments (Rt, Wght) are
//                  register-resident across all 2048 steps.

#define Tn 2048
#define Bn 32
#define In 512
#define Hn 512
#define H3 1536

typedef __attribute__((ext_vector_type(16))) __bf16 v16bf;
typedef __attribute__((ext_vector_type(8)))  __bf16 v8bf;
typedef __attribute__((ext_vector_type(8)))  float  v8f;

union AccU { v8f v; float f[8]; };

__device__ __forceinline__ v8f vzero8() {
  v8f z = {0.f, 0.f, 0.f, 0.f, 0.f, 0.f, 0.f, 0.f};
  return z;
}

// Load one 16x32 bf16 WMMA fragment (A-style; B uses the N-major transposed
// weight so the same contiguous-K load works). src is row-major with leading
// dimension ld (elements). Per the CDNA5 16-bit operand layout:
//   lanes 0-15  : row rbase+lane,    K = kbase+{0..7} and kbase+{16..23}
//   lanes 16-31 : row rbase+lane-16, K = kbase+{8..15} and kbase+{24..31}
// Both chunks are 16B-aligned contiguous loads.
__device__ __forceinline__ v16bf load_frag(const __bf16* src, int ld,
                                           int rbase, int kbase, int lane) {
  int hl = (lane >> 4) & 1;
  int r  = lane & 15;
  const __bf16* p = src + (long)(rbase + r) * ld + kbase + hl * 8;
  v8bf lo = *(const v8bf*)(p);
  v8bf hi = *(const v8bf*)(p + 16);
  return __builtin_shufflevector(lo, hi, 0, 1, 2, 3, 4, 5, 6, 7,
                                 8, 9, 10, 11, 12, 13, 14, 15);
}

__device__ __forceinline__ v8f wmma_bf16(v16bf a, v16bf b, v8f c) {
  return __builtin_amdgcn_wmma_f32_16x16x32_bf16(
      /*neg_a=*/false, a, /*neg_b=*/false, b,
      /*c_mod=*/(short)0, c, /*reuse_a=*/false, /*reuse_b=*/false);
}

// ---------------------------------------------------------------- prep ----

__global__ void k_init(unsigned* cnt, __bf16* hb, const float* h0) {
  int i = blockIdx.x * blockDim.x + threadIdx.x;
  if (i == 0) *cnt = 0u;
  for (int j = i; j < Bn * Hn; j += gridDim.x * blockDim.x) {
    hb[j]            = (__bf16)h0[j];
    hb[Bn * Hn + j]  = (__bf16)0.0f;
  }
}

// src: (K,N) row-major fp32  ->  dst: (N,K) row-major bf16
__global__ void k_transpose(const float* __restrict__ src,
                            __bf16* __restrict__ dst, int K, int N) {
  long total = (long)K * N;
  for (long i = (long)blockIdx.x * blockDim.x + threadIdx.x; i < total;
       i += (long)gridDim.x * blockDim.x) {
    long n = i / K, k = i - n * K;          // coalesced dst writes
    dst[n * K + k] = (__bf16)src[k * N + n];
  }
}

__global__ void k_cvt_x(const float* __restrict__ src,
                        __bf16* __restrict__ dst, long n) {
  for (long i = (long)blockIdx.x * blockDim.x + threadIdx.x; i < n;
       i += (long)gridDim.x * blockDim.x)
    dst[i] = (__bf16)src[i];
}

// ------------------------------------------------- input projection GEMM --
// Grid: (512, 32), 256 threads = 8 wave32.
// Each wave computes a 16(M) x 64(N) strip; M rows = flattened (t,b).
// blockIdx.y 0..23  -> Wx columns (with bx bias), ldo = 1536
// blockIdx.y 24..31 -> Gx columns (no bias),      ldo = 512
__global__ __launch_bounds__(256) void k_inproj(
    const __bf16* __restrict__ xb, const __bf16* __restrict__ Wt,
    const __bf16* __restrict__ Wgxt, const float* __restrict__ bx,
    float* __restrict__ Wx, float* __restrict__ Gx) {
  const int wave = threadIdx.x >> 5;
  const int lane = threadIdx.x & 31;
  const int hl = (lane >> 4) & 1;
  const int r  = lane & 15;
  const int mrow = (blockIdx.x * 8 + wave) * 16;

  const __bf16* wT;
  float* out;
  int ldo, nbase;
  bool hasb;
  if ((int)blockIdx.y < 24) {
    wT = Wt;  out = Wx; ldo = H3; nbase = blockIdx.y * 64; hasb = true;
  } else {
    wT = Wgxt; out = Gx; ldo = Hn; nbase = (blockIdx.y - 24) * 64; hasb = false;
  }

  v8f acc[4];
#pragma unroll
  for (int ns = 0; ns < 4; ++ns) acc[ns] = vzero8();

  for (int kt = 0; kt < In / 32; ++kt) {
    v16bf a = load_frag(xb, In, mrow, kt * 32, lane);
    if (kt + 1 < In / 32)
      __builtin_prefetch(xb + (long)(mrow + r) * In + (kt + 1) * 32, 0, 1);
#pragma unroll
    for (int ns = 0; ns < 4; ++ns) {
      v16bf b = load_frag(wT, In, nbase + ns * 16, kt * 32, lane);
      acc[ns] = wmma_bf16(a, b, acc[ns]);
    }
  }

#pragma unroll
  for (int ns = 0; ns < 4; ++ns) {
    AccU u; u.v = acc[ns];
#pragma unroll
    for (int j = 0; j < 8; ++j) {
      int row = mrow + j + hl * 8;
      int col = nbase + ns * 16 + r;
      float v = u.f[j];
      if (hasb) v += bx[col];
      out[(long)row * ldo + col] = v;
    }
  }
}

// ----------------------------------------------------- recurrent kernel ---
// 32 persistent workgroups x 256 threads (8 wave32). WG `blk` owns hidden
// columns [blk*16, blk*16+16). Double-buffered global h (bf16) + one
// phase-counting grid barrier per timestep. All weight (B-side) fragments
// are loaded ONCE and stay register-resident for all 2048 steps, so the
// per-step dependent chain is: barrier -> 16 independent h-fragment loads
// -> 16 chained WMMAs.
__global__ __launch_bounds__(256) void k_recurrent(
    const float* __restrict__ Wx, const float* __restrict__ Gx,
    const __bf16* __restrict__ Rt, const __bf16* __restrict__ Wght,
    const float* __restrict__ br, const float* __restrict__ bg,
    const float* __restrict__ h0, __bf16* __restrict__ hb,
    unsigned* __restrict__ cnt, float* __restrict__ y,
    float* __restrict__ hn_out, int nblocks) {
  __shared__ float hown[Bn * 16];               // owned fp32 h slice
  __shared__ float zpre[Bn * 16];
  __shared__ float rpre[Bn * 16];
  __shared__ float rhg[Bn * 16];                // recurrent part of g (+br)
  __shared__ float gacc[2][Bn * 16];            // gate GEMM K-split partials

  const int tid  = threadIdx.x;
  const int wave = tid >> 5;
  const int lane = tid & 31;
  const int hl   = (lane >> 4) & 1;
  const int r    = lane & 15;
  const int blk  = blockIdx.x;
  const int colbase = blk * 16;

  // ---- role constants ----
  const int mt  = wave & 1;        // batch tile (rows 0-15 / 16-31)
  const int grp = wave >> 1;       // rh: 0=z, 1=r, 2=g   (waves 0..5)
  const int kh  = wave >> 1;       // gate: K half         (waves 0..3)
  const int nb  = grp * Hn + colbase;

  // ---- register-resident weight fragments (live across all timesteps) ----
  v16bf rfrag[Hn / 32];            // Rt   tile: 16 frags = 128 VGPRs
  v16bf gfrag[8];                  // Wght tile:  8 frags =  64 VGPRs
  if (wave < 6) {
#pragma unroll
    for (int kt = 0; kt < Hn / 32; ++kt)
      rfrag[kt] = load_frag(Rt, Hn, nb, kt * 32, lane);
  }
  if (wave < 4) {
#pragma unroll
    for (int kk = 0; kk < 8; ++kk)
      gfrag[kk] = load_frag(Wght, Hn, colbase, (kh * 8 + kk) * 32, lane);
  }

  for (int i = tid; i < Bn * 16; i += 256) {
    int b = i >> 4, j = i & 15;
    hown[i] = h0[b * Hn + colbase + j];
  }
  __syncthreads();

  for (int t = 0; t < Tn; ++t) {
    const float* wxT = Wx + (long)t * Bn * H3;
    const float* gxT = Gx + (long)t * Bn * Hn;
    const __bf16* hcur  = hb + (t & 1) * (Bn * Hn);
    __bf16*       hnext = hb + ((t + 1) & 1) * (Bn * Hn);

    // ---- recurrent GEMM: rh = h @ R (+br), 6 tiles over waves 0..5 ----
    if (wave < 6) {
      v8f acc = vzero8();
#pragma unroll
      for (int kt = 0; kt < Hn / 32; ++kt) {
        v16bf a = load_frag(hcur, Hn, mt * 16, kt * 32, lane);
        acc = wmma_bf16(a, rfrag[kt], acc);
      }
      float* dst = (grp == 0) ? zpre : (grp == 1) ? rpre : rhg;
      AccU u; u.v = acc;
#pragma unroll
      for (int j = 0; j < 8; ++j) {
        int row = mt * 16 + j + hl * 8;      // batch index
        int col = nb + r;                    // column in 3H
        float v = u.f[j] + br[col];
        if (grp < 2) v += wxT[(long)row * H3 + col];  // wx_z / wx_r
        dst[row * 16 + r] = v;
      }
    }
    __syncthreads();

    // ---- elementwise GRU update on owned slice ----
    for (int i = tid; i < Bn * 16; i += 256) {
      int b = i >> 4, j = i & 15;
      int col = colbase + j;
      float z  = 1.f / (1.f + __expf(-zpre[i]));
      float rr = 1.f / (1.f + __expf(-rpre[i]));
      float g  = tanhf(wxT[(long)b * H3 + 2 * Hn + col] + rr * rhg[i]);
      float hnv = z * hown[i] + (1.f - z) * g;
      hown[i] = hnv;
      hnext[b * Hn + col] = (__bf16)hnv;
    }
    __threadfence();
    __syncthreads();

    // ---- one grid barrier per step (phase-counting, re-zeroed per launch)
    if (tid == 0) {
      __hip_atomic_fetch_add(cnt, 1u, __ATOMIC_RELEASE,
                             __HIP_MEMORY_SCOPE_AGENT);
      unsigned target = (unsigned)(t + 1) * (unsigned)nblocks;
      while (__hip_atomic_load(cnt, __ATOMIC_ACQUIRE,
                               __HIP_MEMORY_SCOPE_AGENT) < target)
        __builtin_amdgcn_s_sleep(1);
    }
    __syncthreads();

    // ---- gate GEMM: h_new @ Wg_h, K split over waves 0..3 ----
    if (wave < 4) {
      v8f acc = vzero8();
#pragma unroll
      for (int kk = 0; kk < 8; ++kk) {
        int kt = kh * 8 + kk;
        v16bf a = load_frag(hnext, Hn, mt * 16, kt * 32, lane);
        acc = wmma_bf16(a, gfrag[kk], acc);
      }
      AccU u; u.v = acc;
#pragma unroll
      for (int j = 0; j < 8; ++j) {
        int row = mt * 16 + j + hl * 8;
        gacc[kh][row * 16 + r] = u.f[j];
      }
    }
    __syncthreads();

    // ---- SiLU gate + output ----
    float* yt = y + (long)t * Bn * Hn;
    for (int i = tid; i < Bn * 16; i += 256) {
      int b = i >> 4, j = i & 15;
      int col = colbase + j;
      float gp = gacc[0][i] + gacc[1][i] + gxT[b * Hn + col] + bg[col];
      float gate = gp * (1.f / (1.f + __expf(-gp)));
      yt[b * Hn + col] = hown[i] * gate;
    }
    __syncthreads();   // protect LDS pre-buffers before next step
  }

  for (int i = tid; i < Bn * 16; i += 256) {
    int b = i >> 4, j = i & 15;
    hn_out[b * Hn + colbase + j] = hown[i];
  }
}

// ------------------------------------------------------------- launcher ---

extern "C" void kernel_launch(void* const* d_in, const int* in_sizes, int n_in,
                              void* d_out, int out_size, void* d_ws,
                              size_t ws_size, hipStream_t stream) {
  (void)in_sizes; (void)n_in; (void)out_size; (void)ws_size;

  const float* x    = (const float*)d_in[0];   // (T,B,I)
  const float* h0   = (const float*)d_in[1];   // (B,H)
  const float* W    = (const float*)d_in[2];   // (I,3H)
  const float* R    = (const float*)d_in[3];   // (H,3H)
  const float* bx   = (const float*)d_in[4];   // (3H)
  const float* br   = (const float*)d_in[5];   // (3H)
  const float* Wg_x = (const float*)d_in[6];   // (I,H)
  const float* Wg_h = (const float*)d_in[7];   // (H,H)
  const float* bg   = (const float*)d_in[8];   // (H)

  float* y      = (float*)d_out;                       // (T,B,H)
  float* hn_out = (float*)d_out + (long)Tn * Bn * Hn;  // (1,B,H)

  // Workspace layout (all offsets 256B-aligned)
  char* ws = (char*)d_ws;
  unsigned* cnt = (unsigned*)(ws + 0);
  __bf16* hb    = (__bf16*)(ws + 256);                        // 2*B*H bf16
  __bf16* Wt    = (__bf16*)(ws + 65792);                      // 3H x I
  __bf16* Wgxt  = (__bf16*)(ws + 1638656);                    // H  x I
  __bf16* Rt    = (__bf16*)(ws + 2162944);                    // 3H x H
  __bf16* Wght  = (__bf16*)(ws + 3735808);                    // H  x H
  __bf16* xb    = (__bf16*)(ws + 4260096);                    // T*B x I
  float*  Wx    = (float*)(ws + 71368960);                    // T*B x 3H
  float*  Gx    = (float*)(ws + 474022144);                   // T*B x H

  k_init<<<64, 256, 0, stream>>>(cnt, hb, h0);
  k_transpose<<<1024, 256, 0, stream>>>(W, Wt, In, H3);
  k_transpose<<<1024, 256, 0, stream>>>(Wg_x, Wgxt, In, Hn);
  k_transpose<<<1024, 256, 0, stream>>>(R, Rt, Hn, H3);
  k_transpose<<<1024, 256, 0, stream>>>(Wg_h, Wght, Hn, Hn);
  k_cvt_x<<<4096, 256, 0, stream>>>(x, xb, (long)Tn * Bn * In);

  dim3 gg(512, 32);
  k_inproj<<<gg, 256, 0, stream>>>(xb, Wt, Wgxt, bx, Wx, Gx);

  k_recurrent<<<32, 256, 0, stream>>>(Wx, Gx, Rt, Wght, br, bg, h0, hb, cnt,
                                      y, hn_out, 32);
}
